// TransformerModel_51642686767197
// MI455X (gfx1250) — matrix operand order repaired and checked by reading, not compiled, not run on hardware
//
#include <hip/hip_runtime.h>
#include <math.h>

#define B_   2
#define S_   1024
#define H_   768
#define NH_  12
#define DH_  64
#define FF_  3072
#define L_   4
#define BS_  (B_*S_)
#define LN_EPS 1e-12f

typedef __attribute__((ext_vector_type(16))) __bf16 bf16x16;
typedef __attribute__((ext_vector_type(8)))  float  f32x8;

// ---------------------------------------------------------------------------
// helpers
// ---------------------------------------------------------------------------
static __device__ inline bf16x16 load2x16B(const void* p0, const void* p1) {
  union { bf16x16 v; uint4 u[2]; } r;
  r.u[0] = *(const uint4*)p0;
  r.u[1] = *(const uint4*)p1;
  return r.v;
}

static __device__ inline f32x8 wmma_bf16(bf16x16 a, bf16x16 b, f32x8 c) {
  // D = A(16x32 bf16) * B(32x16 bf16) + C(16x16 f32)
  return __builtin_amdgcn_wmma_f32_16x16x32_bf16(false, a, false, b, (short)0, c,
                                                 false, false);
}

// async 16B/lane global -> LDS copy (VDST = per-lane LDS byte offset)
static __device__ inline void async_g2l_b128(unsigned lds_off, const void* gaddr) {
  asm volatile("global_load_async_to_lds_b128 %0, %1, off"
               :: "v"(lds_off), "v"(gaddr) : "memory");
}

// ---------------------------------------------------------------------------
// weight packing: f32 row-major [K][N] -> bf16 WMMA-B tiles
// tile (kt,nt) covers K=[kt*32,kt*32+32), N=[nt*16,nt*16+16)
// flat element idx = ((nt*KT + kt)*32 + lane)*16 + i
//   lane<16 : N = nt*16+lane,    K = kt*32 + (i&7) + 16*(i>>3)
//   lane>=16: N = nt*16+lane-16, K = kt*32 + 8 + (i&7) + 16*(i>>3)
// ---------------------------------------------------------------------------
__global__ void pack_w(const float* __restrict__ W, __bf16* __restrict__ out,
                       int K, int N) {
  int idx = blockIdx.x * blockDim.x + threadIdx.x;
  if (idx >= K * N) return;
  int i    = idx & 15;
  int lane = (idx >> 4) & 31;
  int tile = idx >> 9;
  int KT = K >> 5;
  int kt = tile % KT;
  int nt = tile / KT;
  int n = nt * 16 + (lane & 15);
  int k = kt * 32 + ((lane >> 4) ? 8 : 0) + (i & 7) + ((i >> 3) << 4);
  out[idx] = (__bf16)W[(size_t)k * N + n];
}

__global__ void f32_to_bf16(const float* __restrict__ in, __bf16* __restrict__ out,
                            int n) {
  int i = blockIdx.x * blockDim.x + threadIdx.x;
  if (i < n) out[i] = (__bf16)in[i];
}

// ---------------------------------------------------------------------------
// GEMM: C[M,N] = A[M,K](bf16 row-major) @ Wp(packed bf16) + bias
// block (256 thr, 8 waves) -> 256 rows x 64 cols; wave -> 32 rows x 64 cols
// (2 M-tiles x 4 N-tiles = 8 WMMA per 32-wide K-step).
// Packed-B 4KB/K-step staged in LDS via async global->LDS copies (ASYNCcnt),
// double buffered: issue kt+1, wait own slice, barrier, consume kt.
// MODE 0: f32 row-major      MODE 1: bf16 row-major
// MODE 2: gelu -> bf16       MODE 3: bf16 transposed per-head ([b][n][s], V^T)
// ---------------------------------------------------------------------------
template <int MODE>
__global__ void gemm_bf16(const __bf16* __restrict__ A, const __bf16* __restrict__ Wp,
                          const float* __restrict__ bias, void* __restrict__ out,
                          int M, int N, int K) {
  __shared__ __align__(16) __bf16 bstage[2][4 * 512];  // 2 x 4KB

  int tid  = threadIdx.x;
  int wv   = tid >> 5;
  int lane = tid & 31;
  int n4 = N >> 6;
  int bm = blockIdx.x / n4;           // 256-row block
  int ng = blockIdx.x - bm * n4;      // 64-col group
  int KT = K >> 5;
  int halfsel = lane >> 4;
  int lx = lane & 15;
  int aoff = halfsel ? 8 : 0;

  // async staging slice: thread copies 16B = 8 bf16 of the 4KB B block
  int tt  = tid >> 6;                 // which of the 4 N-tiles (64 thr each)
  int tel = (tid & 63) * 8;           // element offset within the 512-elem tile
  const __bf16* wsrc = Wp + ((size_t)(ng * 4 + tt) * KT) * 512 + tel;
  unsigned lds0 = (unsigned)(size_t)&bstage[0][tt * 512 + tel];
  unsigned lds1 = (unsigned)(size_t)&bstage[1][tt * 512 + tel];

  f32x8 acc[2][4];
#pragma unroll
  for (int t = 0; t < 4; ++t) {
    float bv = bias[ng * 64 + t * 16 + lx];
#pragma unroll
    for (int e = 0; e < 8; ++e) { acc[0][t][e] = bv; acc[1][t][e] = bv; }
  }

  const __bf16* arow0 = A + (size_t)(bm * 256 + wv * 32 + lx) * K;
  const __bf16* arow1 = arow0 + (size_t)16 * K;

  async_g2l_b128(lds0, wsrc);  // prologue: stage kt=0

  for (int kt = 0; kt < KT; ++kt) {
    int cur = kt & 1;
    if (kt + 1 < KT) {
      async_g2l_b128(((kt + 1) & 1) ? lds1 : lds0, wsrc + (size_t)(kt + 1) * 512);
      asm volatile("s_wait_asynccnt 0x1" ::: "memory");
    } else {
      asm volatile("s_wait_asynccnt 0x0" ::: "memory");
    }
    __syncthreads();  // all waves' slices of buffer `cur` visible

    const __bf16* ap0 = arow0 + kt * 32 + aoff;
    const __bf16* ap1 = arow1 + kt * 32 + aoff;
    bf16x16 a0 = load2x16B(ap0, ap0 + 16);
    bf16x16 a1 = load2x16B(ap1, ap1 + 16);
    const __bf16* bb = bstage[cur];
#pragma unroll
    for (int t = 0; t < 4; ++t) {
      const __bf16* bp = bb + t * 512 + lane * 16;
      bf16x16 b = load2x16B(bp, bp + 8);
      acc[0][t] = wmma_bf16(a0, b, acc[0][t]);
      acc[1][t] = wmma_bf16(a1, b, acc[1][t]);
    }
    __syncthreads();  // done reading `cur` before it is re-staged
  }

#pragma unroll
  for (int mm = 0; mm < 2; ++mm) {
#pragma unroll
    for (int t = 0; t < 4; ++t) {
#pragma unroll
      for (int e = 0; e < 8; ++e) {
        int m = bm * 256 + wv * 32 + mm * 16 + e + (halfsel ? 8 : 0);
        int n = ng * 64 + t * 16 + lx;
        float v = acc[mm][t][e];
        if (MODE == 0) {
          ((float*)out)[(size_t)m * N + n] = v;
        } else if (MODE == 1) {
          ((__bf16*)out)[(size_t)m * N + n] = (__bf16)v;
        } else if (MODE == 2) {
          float g = 0.5f * v * (1.0f + erff(v * 0.70710678118654752f));
          ((__bf16*)out)[(size_t)m * N + n] = (__bf16)g;
        } else { // MODE 3: V^T  out[b*H*S + n*S + s]
          int bb2 = m / S_, ss = m % S_;
          ((__bf16*)out)[((size_t)bb2 * H_ + n) * S_ + ss] = (__bf16)v;
        }
      }
    }
  }
}

// ---------------------------------------------------------------------------
// flash attention: 1 wave per (b, head, 16-query block); keys in blocks of 32
// ---------------------------------------------------------------------------
__global__ void attention(const __bf16* __restrict__ q, const __bf16* __restrict__ k,
                          const __bf16* __restrict__ vt, const int* __restrict__ mask,
                          __bf16* __restrict__ ctx) {
  __shared__ __align__(16) __bf16 pl[16 * 32];  // P tile, row-major [16 q][32 key]
  int lane = threadIdx.x;
  int qb = blockIdx.x & 63;                     // S/16 = 64
  int h  = (blockIdx.x >> 6) % NH_;
  int b  = blockIdx.x / (64 * NH_);
  int halfsel = lane >> 4;
  int lx = lane & 15;
  int aoff = halfsel ? 8 : 0;

  // Q A-operand tiles for DH=64 (two 32-wide k-steps)
  const __bf16* qrow = q + (size_t)(b * S_ + qb * 16 + lx) * H_ + h * DH_;
  bf16x16 qa0 = load2x16B(qrow + aoff,      qrow + aoff + 16);
  bf16x16 qa1 = load2x16B(qrow + 32 + aoff, qrow + 32 + aoff + 16);

  float mrun[8], lrun[8];
  f32x8 O[4];
#pragma unroll
  for (int e = 0; e < 8; ++e) { mrun[e] = -INFINITY; lrun[e] = 0.f; }
#pragma unroll
  for (int t = 0; t < 4; ++t)
#pragma unroll
    for (int e = 0; e < 8; ++e) O[t][e] = 0.f;

  const float scale = 0.125f;                   // 1/sqrt(64)
  const int* mrow = mask + (size_t)b * S_ * S_;
  const __bf16* vbase = vt + ((size_t)b * H_ + h * DH_) * S_;

  for (int kb = 0; kb < S_; kb += 32) {
    float st[2][8];
#pragma unroll
    for (int sub = 0; sub < 2; ++sub) {
      f32x8 sacc;
#pragma unroll
      for (int e = 0; e < 8; ++e) sacc[e] = 0.f;
      int keyrow = kb + sub * 16 + lx;
      const __bf16* kp = k + (size_t)(b * S_ + keyrow) * H_ + h * DH_ + (halfsel ? 8 : 0);
      bf16x16 kb0 = load2x16B(kp, kp + 16);         // dh 0..31 slice
      bf16x16 kb1 = load2x16B(kp + 32, kp + 48);    // dh 32..63 slice
      sacc = wmma_bf16(qa0, kb0, sacc);
      sacc = wmma_bf16(qa1, kb1, sacc);
#pragma unroll
      for (int e = 0; e < 8; ++e) {
        int qg = qb * 16 + e + (halfsel ? 8 : 0);
        int kg = kb + sub * 16 + lx;
        float sc = sacc[e] * scale;
        sc += (1.0f - (float)mrow[(size_t)qg * S_ + kg]) * -10000.0f;
        st[sub][e] = sc;
      }
    }

    // online softmax (row = 16 lanes within a half-wave, C-layout)
    float pvals[2][8];
#pragma unroll
    for (int e = 0; e < 8; ++e) {
      float tmax = fmaxf(st[0][e], st[1][e]);
      for (int off = 1; off < 16; off <<= 1) tmax = fmaxf(tmax, __shfl_xor(tmax, off));
      float mnew = fmaxf(mrun[e], tmax);
      float corr = __expf(mrun[e] - mnew);
      lrun[e] *= corr;
#pragma unroll
      for (int t = 0; t < 4; ++t) O[t][e] *= corr;
      float p0 = __expf(st[0][e] - mnew);
      float p1 = __expf(st[1][e] - mnew);
      pvals[0][e] = p0; pvals[1][e] = p1;
      float rs = p0 + p1;
      for (int off = 1; off < 16; off <<= 1) rs += __shfl_xor(rs, off);
      lrun[e] += rs;
      mrun[e] = mnew;
    }

    // C-layout f32 -> A-layout bf16 via LDS bounce
#pragma unroll
    for (int sub = 0; sub < 2; ++sub)
#pragma unroll
      for (int e = 0; e < 8; ++e)
        pl[(e + (halfsel ? 8 : 0)) * 32 + lx + sub * 16] = (__bf16)pvals[sub][e];
    __syncthreads();
    const __bf16* pbase = pl + lx * 32 + aoff;
    bf16x16 pa = load2x16B(pbase, pbase + 16);

    // O += P(16x32) @ V(32x64): 4 WMMAs, V^T rows are contiguous B-tiles
#pragma unroll
    for (int t = 0; t < 4; ++t) {
      const __bf16* vp = vbase + (size_t)(t * 16 + lx) * S_ + kb + (halfsel ? 8 : 0);
      bf16x16 bv = load2x16B(vp, vp + 16);
      O[t] = wmma_bf16(pa, bv, O[t]);
    }
    __syncthreads();
  }

  // normalize and store ctx bf16 row-major [BS][H]
#pragma unroll
  for (int e = 0; e < 8; ++e) {
    float inv = 1.0f / lrun[e];
    int m = qb * 16 + e + (halfsel ? 8 : 0);
    __bf16* crow = ctx + (size_t)(b * S_ + m) * H_ + h * DH_;
#pragma unroll
    for (int t = 0; t < 4; ++t)
      crow[t * 16 + lx] = (__bf16)(O[t][e] * inv);
  }
}

// ---------------------------------------------------------------------------
// residual add + LayerNorm; one wave per row; emits f32 + bf16 copies
// ---------------------------------------------------------------------------
__global__ void residual_layernorm(const float* __restrict__ proj,
                                   const float* __restrict__ resid,
                                   const float* __restrict__ g,
                                   const float* __restrict__ bta,
                                   float* __restrict__ out_f32,
                                   __bf16* __restrict__ out_bf16) {
  int wid  = (blockIdx.x * blockDim.x + threadIdx.x) >> 5;
  int lane = threadIdx.x & 31;
  constexpr int NPL = H_ / 32;  // 24
  size_t base = (size_t)wid * H_;
  float v[NPL];
  float s = 0.f;
#pragma unroll
  for (int j = 0; j < NPL; ++j) {
    int c = lane + j * 32;
    v[j] = proj[base + c] + resid[base + c];
    s += v[j];
  }
  for (int off = 16; off > 0; off >>= 1) s += __shfl_xor(s, off);
  float mean = s * (1.0f / H_);
  float sq = 0.f;
#pragma unroll
  for (int j = 0; j < NPL; ++j) { float d = v[j] - mean; sq += d * d; }
  for (int off = 16; off > 0; off >>= 1) sq += __shfl_xor(sq, off);
  float rstd = rsqrtf(sq * (1.0f / H_) + LN_EPS);
#pragma unroll
  for (int j = 0; j < NPL; ++j) {
    int c = lane + j * 32;
    float y = (v[j] - mean) * rstd * g[c] + bta[c];
    out_f32[base + c] = y;
    out_bf16[base + c] = (__bf16)y;
  }
}

// ---------------------------------------------------------------------------
// host launcher
// ---------------------------------------------------------------------------
extern "C" void kernel_launch(void* const* d_in, const int* in_sizes, int n_in,
                              void* d_out, int out_size, void* d_ws, size_t ws_size,
                              hipStream_t stream) {
  const float* x    = (const float*)d_in[0];
  const int*   mask = (const int*)d_in[1];
  const float* Wq = (const float*)d_in[2];  const float* bq = (const float*)d_in[3];
  const float* Wk = (const float*)d_in[4];  const float* bk = (const float*)d_in[5];
  const float* Wv = (const float*)d_in[6];  const float* bv = (const float*)d_in[7];
  const float* Wao= (const float*)d_in[8];  const float* bao= (const float*)d_in[9];
  const float* g1 = (const float*)d_in[10]; const float* b1 = (const float*)d_in[11];
  const float* Wi = (const float*)d_in[12]; const float* bi = (const float*)d_in[13];
  const float* Wfo= (const float*)d_in[14]; const float* bfo= (const float*)d_in[15];
  const float* g2 = (const float*)d_in[16]; const float* b2 = (const float*)d_in[17];

  char* ws = (char*)d_ws;
  size_t off = 0;
  auto alloc = [&](size_t bytes) {
    void* p = ws + off;
    off += (bytes + 255) & ~(size_t)255;
    return p;
  };
  const size_t HH = (size_t)H_ * H_, HF = (size_t)H_ * FF_;
  const size_t PWL = 4 * HH + 2 * HF;  // packed bf16 elements per layer

  __bf16* pw     = (__bf16*)alloc((size_t)L_ * PWL * 2);
  float*  hbuf   = (float*) alloc((size_t)BS_ * H_ * 4);
  __bf16* abuf   = (__bf16*)alloc((size_t)BS_ * H_ * 2);
  __bf16* qbuf   = (__bf16*)alloc((size_t)BS_ * H_ * 2);
  __bf16* kbuf   = (__bf16*)alloc((size_t)BS_ * H_ * 2);
  __bf16* vtbuf  = (__bf16*)alloc((size_t)BS_ * H_ * 2);
  __bf16* ctxbuf = (__bf16*)alloc((size_t)BS_ * H_ * 2);
  float*  projf  = (float*) alloc((size_t)BS_ * H_ * 4);
  float*  attnf  = (float*) alloc((size_t)BS_ * H_ * 4);
  __bf16* attnbf = (__bf16*)alloc((size_t)BS_ * H_ * 2);
  __bf16* interbf= (__bf16*)alloc((size_t)BS_ * FF_ * 2);

  // pack all layer weights into WMMA-B tile layout (bf16)
  for (int l = 0; l < L_; ++l) {
    __bf16* base = pw + (size_t)l * PWL;
    auto packs = [&](const float* src, __bf16* dst, int K, int N) {
      int n = K * N;
      pack_w<<<(n + 255) / 256, 256, 0, stream>>>(src, dst, K, N);
    };
    packs(Wq  + l * HH, base,              H_,  H_);
    packs(Wk  + l * HH, base + HH,         H_,  H_);
    packs(Wv  + l * HH, base + 2 * HH,     H_,  H_);
    packs(Wao + l * HH, base + 3 * HH,     H_,  H_);
    packs(Wi  + l * HF, base + 4 * HH,     H_,  FF_);
    packs(Wfo + l * HF, base + 4 * HH + HF, FF_, H_);
  }

  hipMemcpyAsync(hbuf, x, (size_t)BS_ * H_ * 4, hipMemcpyDeviceToDevice, stream);
  f32_to_bf16<<<(BS_ * H_ + 255) / 256, 256, 0, stream>>>(x, abuf, BS_ * H_);

  const int gH = (BS_ / 256) * (H_ / 64);   // 96 blocks (8 waves each)
  const int gF = (BS_ / 256) * (FF_ / 64);  // 384 blocks

  for (int l = 0; l < L_; ++l) {
    __bf16* base = pw + (size_t)l * PWL;
    gemm_bf16<1><<<gH, 256, 0, stream>>>(abuf,   base,              bq + l * H_,  qbuf,  BS_, H_,  H_);
    gemm_bf16<1><<<gH, 256, 0, stream>>>(abuf,   base + HH,         bk + l * H_,  kbuf,  BS_, H_,  H_);
    gemm_bf16<3><<<gH, 256, 0, stream>>>(abuf,   base + 2 * HH,     bv + l * H_,  vtbuf, BS_, H_,  H_);
    attention<<<B_ * NH_ * (S_ / 16), 32, 0, stream>>>(qbuf, kbuf, vtbuf, mask, ctxbuf);
    gemm_bf16<0><<<gH, 256, 0, stream>>>(ctxbuf, base + 3 * HH,     bao + l * H_, projf, BS_, H_,  H_);
    residual_layernorm<<<BS_ / 8, 256, 0, stream>>>(projf, hbuf, g1 + l * H_, b1 + l * H_, attnf, attnbf);
    gemm_bf16<2><<<gF, 256, 0, stream>>>(attnbf, base + 4 * HH,     bi + l * FF_, interbf, BS_, FF_, H_);
    gemm_bf16<0><<<gH, 256, 0, stream>>>(interbf, base + 4 * HH + HF, bfo + l * H_, projf, BS_, H_, FF_);
    float* outp = (l == L_ - 1) ? (float*)d_out : hbuf;
    residual_layernorm<<<BS_ / 8, 256, 0, stream>>>(projf, attnf, g2 + l * H_, b2 + l * H_, outp, abuf);
  }
  (void)in_sizes; (void)n_in; (void)out_size; (void)ws_size;
}
